// GraphVAE_17377437680240
// MI455X (gfx1250) — matrix operand order repaired and checked by reading, not compiled, not run on hardware
//
#include <hip/hip_runtime.h>
#include <hip/hip_bf16.h>
#include <math.h>

// ---------------------------------------------------------------------------
// GraphVAE forward for MI455X (gfx1250, wave32, WMMA).
//  * All GEMMs: v_wmma_f32_16x16x32_bf16, split-precision bf16 hi/lo (3 WMMAs)
//    for ~fp32 accuracy; operands pre-packed once (A row-major, B transposed)
//    so every fragment is a contiguous 16/32B load in one contiguous k order shared by A and B.
//  * A_pred = sigmoid(z z^T): symmetric -> compute lower-triangular 32x32
//    blocks only (2x2 WMMA tiles per wave), store block + mirrored block.
//  * GCN aggregation: degree-normalized scatter-sum via f32 global atomics.
// ---------------------------------------------------------------------------

typedef __bf16 bf16;
typedef bf16  v16bf __attribute__((ext_vector_type(16)));
typedef bf16  v8bf  __attribute__((ext_vector_type(8)));
typedef float v8f   __attribute__((ext_vector_type(8)));

#define NNODES 10000
#define NEDGES 320000
#define TILES  625          // 10000 / 16

// ---------------- degree / normalization ----------------------------------

__global__ void k_deg_init(float* __restrict__ deg, int n) {
  int i = blockIdx.x * blockDim.x + threadIdx.x;
  if (i < n) deg[i] = 1.0f;  // self-loop contributes 1
}

__global__ void k_deg_count(const int* __restrict__ dst, float* __restrict__ deg, int e) {
  int i = blockIdx.x * blockDim.x + threadIdx.x;
  if (i < e) atomicAdd(&deg[dst[i]], 1.0f);
}

__global__ void k_deg_rsqrt(float* __restrict__ deg, int n) {
  int i = blockIdx.x * blockDim.x + threadIdx.x;
  if (i < n) deg[i] = rsqrtf(deg[i]);  // deg >= 1 always
}

// ---------------- fp32 -> bf16 hi/lo pre-pack ------------------------------

__global__ void k_pack(const float* __restrict__ X, bf16* __restrict__ H,
                       bf16* __restrict__ L, int n) {
  int i = blockIdx.x * blockDim.x + threadIdx.x;
  if (i >= n) return;
  float f = X[i];
  bf16 h = (bf16)f;
  H[i] = h;
  L[i] = (bf16)(f - (float)h);
}

// Pack W[K][N] transposed into Bt[N][K] bf16 hi/lo (weights are tiny).
__global__ void k_pack_t(const float* __restrict__ W, bf16* __restrict__ H,
                         bf16* __restrict__ L, int K, int N) {
  int i = blockIdx.x * blockDim.x + threadIdx.x;
  if (i >= K * N) return;
  int k = i / N;
  int n = i - k * N;
  float f = W[i];
  bf16 h = (bf16)f;
  size_t o = (size_t)n * K + k;
  H[o] = h;
  L[o] = (bf16)(f - (float)h);
}

// ---------------- WMMA fragment helpers ------------------------------------

__device__ __forceinline__ v16bf cat8(v8bf a, v8bf b) {
  return __builtin_shufflevector(a, b, 0,1,2,3,4,5,6,7,8,9,10,11,12,13,14,15);
}

// A-matrix 16x32 bf16 fragment, this code's contiguous k order (not ISA table order): for this lane's row,
// elements 0..15 = p[0..15]
// (p = row base + k0 + half*16; covers K {0-15} / {16-31}).
__device__ __forceinline__ v16bf load_a_frag(const bf16* __restrict__ p) {
  v8bf lo = *(const v8bf*)p;
  v8bf hi = *(const v8bf*)(p + 8);
  return cat8(lo, hi);
}

__device__ __forceinline__ v8f wmma3(v16bf ah, v16bf al, v16bf bh, v16bf bl, v8f c) {
  c = __builtin_amdgcn_wmma_f32_16x16x32_bf16(false, ah, false, bh, (short)0, c, false, false);
  c = __builtin_amdgcn_wmma_f32_16x16x32_bf16(false, ah, false, bl, (short)0, c, false, false);
  c = __builtin_amdgcn_wmma_f32_16x16x32_bf16(false, al, false, bh, (short)0, c, false, false);
  return c;
}

// ---------------- generic GEMM: C = A[M,K] @ B[K,N] ------------------------
// A pre-packed bf16 hi/lo row-major [M][K]; B pre-packed TRANSPOSED [N][K].
// B-fragment: lane n element e = B[k0+half*16+e][n] = Bt[n][k0+half*16+e]
// -> one contiguous 32B v16bf load per fragment.
// One wave per 16x16 tile; M,N multiples of 16, K multiple of 32.

__global__ void __launch_bounds__(256)
k_gemm_bf(const bf16* __restrict__ Ah, const bf16* __restrict__ Al,
          const bf16* __restrict__ Bth, const bf16* __restrict__ Btl,
          float* __restrict__ C, int M, int N, int K) {
  int wave = blockIdx.x * (blockDim.x >> 5) + (threadIdx.x >> 5);
  int tilesN = N >> 4;
  if (wave >= (M >> 4) * tilesN) return;   // wave-uniform: EXEC stays all-ones
  int lane = threadIdx.x & 31;
  int half = lane >> 4;
  int l    = lane & 15;
  int tM = wave / tilesN;
  int tN = wave - tM * tilesN;
  int row = tM * 16 + l;
  int col = tN * 16 + l;

  v8f acc = {0.f, 0.f, 0.f, 0.f, 0.f, 0.f, 0.f, 0.f};
  for (int k0 = 0; k0 < K; k0 += 32) {
    size_t aoff = (size_t)row * K + k0 + half * 16;
    v16bf ah = load_a_frag(Ah + aoff);
    v16bf al = load_a_frag(Al + aoff);
    size_t boff = (size_t)col * K + k0 + half * 16;
    v16bf bh = *(const v16bf*)(Bth + boff);
    v16bf bl = *(const v16bf*)(Btl + boff);
    acc = wmma3(ah, al, bh, bl, acc);
  }

  // C/D: lanes 0-15 -> M=v, lanes 16-31 -> M=8+v; N = l.
  float* cp = C + (size_t)(tM * 16 + half * 8) * N + col;
#pragma unroll
  for (int v = 0; v < 8; ++v) cp[(size_t)v * N] = acc[v];
}

// ---------------- A_pred = sigmoid(Z Z^T), symmetric, blocked 2x2 ----------
// Z pre-packed bf16 hi/lo, [NNODES][64] row-major (rows of Z are both the
// A rows and the B columns). Each wave: one 32x32 lower-triangular block
// (4 accumulators); stores block + mirrored block.

__device__ __forceinline__ v8f sig8(v8f a) {
  v8f r;
#pragma unroll
  for (int v = 0; v < 8; ++v) r[v] = 1.0f / (1.0f + expf(-a[v]));
  return r;
}

__device__ __forceinline__ void store_direct(float* __restrict__ P, v8f s,
                                             int tM, int tN, int half, int l) {
  float* cp = P + (size_t)(tM * 16 + half * 8) * NNODES + tN * 16 + l;
#pragma unroll
  for (int v = 0; v < 8; ++v) cp[(size_t)v * NNODES] = s[v];
}

__device__ __forceinline__ void store_mirror(float* __restrict__ P, v8f s,
                                             int tM, int tN, int half, int l) {
  // element (r0+v, c) -> (c, r0+v): 8 contiguous floats per lane
  float* mp = P + (size_t)(tN * 16 + l) * NNODES + tM * 16 + half * 8;
#pragma unroll
  for (int v = 0; v < 8; ++v) mp[v] = s[v];
}

__global__ void __launch_bounds__(256)
k_zzt_sigmoid(const bf16* __restrict__ Zh, const bf16* __restrict__ Zl,
              float* __restrict__ P, int nblk) {
  int w = blockIdx.x * (blockDim.x >> 5) + (threadIdx.x >> 5);
  if (w >= nblk) return;                   // wave-uniform
  int lane = threadIdx.x & 31;
  int half = lane >> 4;
  int l    = lane & 15;

  // triangular block index: w -> (bm, bn), bm >= bn
  int bm = (int)((sqrtf(8.0f * (float)w + 1.0f) - 1.0f) * 0.5f);
  while ((bm + 1) * (bm + 2) / 2 <= w) ++bm;
  while (bm * (bm + 1) / 2 > w) --bm;
  int bn = w - bm * (bm + 1) / 2;

  bool v1m = (2 * bm + 1 < TILES);
  bool v1n = (2 * bn + 1 < TILES);
  int t0m = 2 * bm, t1m = v1m ? 2 * bm + 1 : 2 * bm;   // clamp for loads
  int t0n = 2 * bn, t1n = v1n ? 2 * bn + 1 : 2 * bn;

  v8f a00 = {0.f,0.f,0.f,0.f,0.f,0.f,0.f,0.f};
  v8f a01 = a00, a10 = a00, a11 = a00;

#pragma unroll
  for (int k0 = 0; k0 < 64; k0 += 32) {
    size_t r0 = (size_t)(t0m * 16 + l) * 64 + k0 + half * 16;
    size_t r1 = (size_t)(t1m * 16 + l) * 64 + k0 + half * 16;
    v16bf a0h = load_a_frag(Zh + r0), a0l = load_a_frag(Zl + r0);
    v16bf a1h = load_a_frag(Zh + r1), a1l = load_a_frag(Zl + r1);
    // B columns are rows of Z: 16 contiguous bf16 (32B aligned)
    size_t c0 = (size_t)(t0n * 16 + l) * 64 + k0 + half * 16;
    size_t c1 = (size_t)(t1n * 16 + l) * 64 + k0 + half * 16;
    v16bf b0h = *(const v16bf*)(Zh + c0), b0l = *(const v16bf*)(Zl + c0);
    v16bf b1h = *(const v16bf*)(Zh + c1), b1l = *(const v16bf*)(Zl + c1);

    a00 = wmma3(a0h, a0l, b0h, b0l, a00);
    a01 = wmma3(a0h, a0l, b1h, b1l, a01);
    a10 = wmma3(a1h, a1l, b0h, b0l, a10);
    a11 = wmma3(a1h, a1l, b1h, b1l, a11);
  }

  v8f s00 = sig8(a00), s01 = sig8(a01), s10 = sig8(a10), s11 = sig8(a11);

  store_direct(P, s00, t0m, t0n, half, l);
  if (v1n)        store_direct(P, s01, t0m, t1n, half, l);
  if (v1m)        store_direct(P, s10, t1m, t0n, half, l);
  if (v1m && v1n) store_direct(P, s11, t1m, t1n, half, l);
  if (bm != bn) {  // off-diagonal block: mirror into the upper triangle
    store_mirror(P, s00, t0m, t0n, half, l);
    if (v1n)        store_mirror(P, s01, t0m, t1n, half, l);
    if (v1m)        store_mirror(P, s10, t1m, t0n, half, l);
    if (v1m && v1n) store_mirror(P, s11, t1m, t1n, half, l);
  }
}

// ---------------- aggregation (scatter-sum) --------------------------------

__global__ void k_agg_init(const float* __restrict__ xw, const float* __restrict__ bias,
                           const float* __restrict__ dinv, float* __restrict__ out,
                           int n, int F) {
  int i = blockIdx.x * blockDim.x + threadIdx.x;
  if (i >= n * F) return;
  int node = i / F;
  int f    = i - node * F;
  float di = dinv[node];
  out[i] = bias[f] + xw[i] * di * di;
}

__global__ void k_agg_scatter(const int* __restrict__ src, const int* __restrict__ dst,
                              const float* __restrict__ xw, const float* __restrict__ dinv,
                              float* __restrict__ out, int e, int F) {
  int i = blockIdx.x * blockDim.x + threadIdx.x;
  int fq = F >> 2;
  int eidx = i / fq;
  if (eidx >= e) return;
  int f4 = (i - eidx * fq) << 2;
  int s = src[eidx];
  int d = dst[eidx];
  float w = dinv[s] * dinv[d];
  const float4 v = *(const float4*)(xw + (size_t)s * F + f4);
  float* op = out + (size_t)d * F + f4;
  atomicAdd(op + 0, v.x * w);
  atomicAdd(op + 1, v.y * w);
  atomicAdd(op + 2, v.z * w);
  atomicAdd(op + 3, v.w * w);
}

// ---------------- elementwise ----------------------------------------------

__global__ void k_relu(float* __restrict__ x, int n) {
  int i = blockIdx.x * blockDim.x + threadIdx.x;
  if (i < n) x[i] = fmaxf(x[i], 0.0f);
}

__global__ void k_sigmoid(float* __restrict__ x, int n) {
  int i = blockIdx.x * blockDim.x + threadIdx.x;
  if (i < n) x[i] = 1.0f / (1.0f + expf(-x[i]));
}

__global__ void k_reparam(const float* __restrict__ eps, const float* __restrict__ mu,
                          const float* __restrict__ lv, float* __restrict__ z, int n) {
  int i = blockIdx.x * blockDim.x + threadIdx.x;
  if (i < n) z[i] = eps[i] * expf(0.5f * lv[i]) + mu[i];
}

// ---------------- launcher --------------------------------------------------

extern "C" void kernel_launch(void* const* d_in, const int* in_sizes, int n_in,
                              void* d_out, int out_size, void* d_ws, size_t ws_size,
                              hipStream_t stream) {
  (void)in_sizes; (void)n_in; (void)out_size; (void)ws_size;

  const float* x   = (const float*)d_in[0];
  const int*   ei  = (const int*)d_in[1];      // [2, E] int32
  const float* eps = (const float*)d_in[2];
  const float* W1  = (const float*)d_in[3];  const float* b1  = (const float*)d_in[4];
  const float* Wmu = (const float*)d_in[5];  const float* bmu = (const float*)d_in[6];
  const float* Wlv = (const float*)d_in[7];  const float* blv = (const float*)d_in[8];
  const float* W2  = (const float*)d_in[9];  const float* b2  = (const float*)d_in[10];
  const float* W3  = (const float*)d_in[11]; const float* b3  = (const float*)d_in[12];

  const int* src = ei;
  const int* dst = ei + NEDGES;

  float* out   = (float*)d_out;
  float* Apred = out;                       // 1e8
  float* mu    = out + 100000000ull;        // 640k
  float* lv    = mu + 640000ull;            // 640k
  float* z     = lv + 640000ull;            // 640k
  float* xpred = z  + 640000ull;            // 5.12M

  // workspace: fp32 scratch + bf16 hi/lo packed operands (all 32B+ aligned)
  float* ws   = (float*)d_ws;
  float* dinv = ws;                         // 10240 f32
  float* bufA = ws + 10240;                 // 10000*512 f32 (XW scratch)
  float* bufH = bufA + 5120000;             // 10000*256 f32 (h / xd)
  bf16*  pAh  = (bf16*)(bufH + 2560000);    // 10000*512 bf16
  bf16*  pAl  = pAh + 5120000;
  bf16*  pBh  = pAl + 5120000;              // 512*256 max weight (transposed)
  bf16*  pBl  = pBh + 131072;
  bf16*  pZh  = pBl + 131072;               // 10000*64
  bf16*  pZl  = pZh + 640000;

  const int TB = 256;
  auto cdiv = [](long long a, long long b) { return (int)((a + b - 1) / b); };

  // --- degrees -> dinv ---
  k_deg_init <<<cdiv(NNODES, TB), TB, 0, stream>>>(dinv, NNODES);
  k_deg_count<<<cdiv(NEDGES, TB), TB, 0, stream>>>(dst, dinv, NEDGES);
  k_deg_rsqrt<<<cdiv(NNODES, TB), TB, 0, stream>>>(dinv, NNODES);

  auto pack = [&](const float* X, bf16* H, bf16* L, long long n) {
    k_pack<<<cdiv(n, TB), TB, 0, stream>>>(X, H, L, (int)n);
  };
  auto gcnP = [&](const bf16* Ah, const bf16* Al, const float* W, const float* b,
                  float* XW, float* Out, int F, int K) {
    k_pack_t<<<cdiv((long long)K * F, TB), TB, 0, stream>>>(W, pBh, pBl, K, F);
    int tiles = (NNODES >> 4) * (F >> 4);
    k_gemm_bf<<<cdiv(tiles, 8), TB, 0, stream>>>(Ah, Al, pBh, pBl, XW, NNODES, F, K);
    k_agg_init<<<cdiv((long long)NNODES * F, TB), TB, 0, stream>>>(XW, b, dinv, Out, NNODES, F);
    k_agg_scatter<<<cdiv((long long)NEDGES * (F >> 2), TB), TB, 0, stream>>>(
        src, dst, XW, dinv, Out, NEDGES, F);
  };

  // --- encoder: h = relu(gcn(x, W1, b1)) ---
  pack(x, pAh, pAl, (long long)NNODES * 512);
  gcnP(pAh, pAl, W1, b1, bufA, bufH, 256, 512);
  k_relu<<<cdiv((long long)NNODES * 256, TB), TB, 0, stream>>>(bufH, NNODES * 256);

  // --- mu / logvar ---
  pack(bufH, pAh, pAl, (long long)NNODES * 256);
  gcnP(pAh, pAl, Wmu, bmu, bufA, mu, 64, 256);
  gcnP(pAh, pAl, Wlv, blv, bufA, lv, 64, 256);

  // --- z = eps * exp(0.5*logvar) + mu ---
  k_reparam<<<cdiv((long long)NNODES * 64, TB), TB, 0, stream>>>(eps, mu, lv, z, NNODES * 64);

  // --- A_pred = sigmoid(z z^T): lower-triangular 32x32 blocks + mirror ---
  pack(z, pZh, pZl, (long long)NNODES * 64);
  {
    const int nbp = (TILES + 1) / 2;            // 313 block rows
    const int nblk = nbp * (nbp + 1) / 2;       // 49141 triangular blocks
    k_zzt_sigmoid<<<cdiv(nblk, 8), TB, 0, stream>>>(pZh, pZl, Apred, nblk);
  }

  // --- decoder: xd = relu(gcn(z, W2, b2)) (z already packed) ---
  gcnP(pZh, pZl, W2, b2, bufA, bufH, 256, 64);
  k_relu<<<cdiv((long long)NNODES * 256, TB), TB, 0, stream>>>(bufH, NNODES * 256);

  // --- x_pred = sigmoid(gcn(xd, W3, b3)) ---
  pack(bufH, pAh, pAl, (long long)NNODES * 256);
  gcnP(pAh, pAl, W3, b3, bufA, xpred, 512, 256);
  k_sigmoid<<<cdiv((long long)NNODES * 512, TB), TB, 0, stream>>>(xpred, NNODES * 512);
}
